// NDT_15040975471043
// MI455X (gfx1250) — matrix operand order repaired and checked by reading, not compiled, not run on hardware
//
#include <hip/hip_runtime.h>
#include <hip/hip_bf16.h>

// ---------------------------------------------------------------- constants
#define KDIM   1024
#define NCOLS  32736          // 32 trees * 1023 forks
#define BATCH  2048
#define TREES  32
#define FORKS  1023
#define EPS_BN 1e-5f
#define EPS_CAP 1e-5f

typedef __bf16 bf16;
typedef __attribute__((ext_vector_type(16))) __bf16 v16bf;
typedef __attribute__((ext_vector_type(8)))  __bf16 v8bf;
typedef __attribute__((ext_vector_type(8)))  float  v8f;

union FragBF { v16bf v; v8bf h[2]; };

// ---------------------------------------------------------------- 1) split f32 -> bf16 hi/lo
__global__ void ndt_split_kernel(const float* __restrict__ x, const float* __restrict__ W,
                                 bf16* __restrict__ xh, bf16* __restrict__ xl,
                                 bf16* __restrict__ wh, bf16* __restrict__ wl)
{
    const size_t NX = (size_t)BATCH * KDIM;     // 2,097,152
    const size_t NW = (size_t)NCOLS * KDIM;     // 33,521,664
    for (size_t i = (size_t)blockIdx.x * blockDim.x + threadIdx.x; i < NW;
         i += (size_t)gridDim.x * blockDim.x) {
        float v = W[i];
        bf16 h = (bf16)v;
        bf16 l = (bf16)(v - (float)h);
        wh[i] = h; wl[i] = l;
        if (i < NX) {
            float u = x[i];
            bf16 uh = (bf16)u;
            bf16 ul = (bf16)(u - (float)uh);
            xh[i] = uh; xl[i] = ul;
        }
    }
}

// ---------------------------------------------------------------- 2) WMMA GEMM, async-LDS double buffered
// C[m,n] = sum_k x[m,k]*W[n,k].  Block = 8 waves (2 M x 4 N), wave tile 32x64,
// macro tile 64x256, K-tile 32.  hi/lo bf16 split: AhBh + AhBl + AlBh (fp32-grade).
// A/B k-tiles staged global->LDS with global_load_async_to_lds_b128 (ASYNCcnt),
// double buffered; fragments fed from ds_load_b128.

#define ASTR 40                      // LDS row stride in halves: 80 B -> bank-conflict free, 16B aligned
#define A_REG  (2 * 64 * ASTR)       // per-buffer A region (hi+lo) = 5120 halves
#define B_REG  (2 * 256 * ASTR)      // per-buffer B region (hi+lo) = 20480 halves
#define SMEM_HALVES (2 * A_REG + 2 * B_REG)   // 51200 halves = 100 KB

__device__ __forceinline__ void async_ld_b128(unsigned lds_byte, const bf16* g)
{
    asm volatile("global_load_async_to_lds_b128 %0, %1, off"
                 :: "v"(lds_byte), "v"((unsigned long long)(uintptr_t)g)
                 : "memory");
}
__device__ __forceinline__ void wait_async0()
{
    asm volatile("s_wait_asynccnt 0" ::: "memory");
}

__global__ __launch_bounds__(256, 1)
void ndt_gemm_kernel(const bf16* __restrict__ xh, const bf16* __restrict__ xl,
                     const bf16* __restrict__ wh, const bf16* __restrict__ wl,
                     bf16* __restrict__ logits)
{
    __shared__ __align__(16) bf16 smem[SMEM_HALVES];

    const int tid   = threadIdx.x;
    const int lane  = tid & 31;
    const int wid   = tid >> 5;
    const int waveM = wid >> 2;            // 0..1
    const int waveN = wid & 3;             // 0..3
    const int mblk  = blockIdx.y * 64;
    const int nblk  = blockIdx.x * 256;

    const unsigned sbase = (unsigned)(uintptr_t)&smem[0];
    // region offsets in halves
    const unsigned oAh0 = 0,              oAl0 = 64 * ASTR;
    const unsigned oBh0 = 2 * A_REG,      oBl0 = 2 * A_REG + 256 * ASTR;

    // staging coordinates (uniform per thread)
    const int sAr = tid >> 2, sAc = (tid & 3) * 8;           // A: 1 chunk each for hi and lo
    int sBr[4], sBc[4], sBcol[4];                            // B: 4 chunks each for hi and lo
#pragma unroll
    for (int t = 0; t < 4; ++t) {
        const int id = tid + 256 * t;
        sBr[t] = id >> 2;
        sBc[t] = (id & 3) * 8;
        const int col = nblk + sBr[t];
        sBcol[t] = (col < NCOLS) ? col : (NCOLS - 1);
    }

    auto stage = [&](int buf, int kk) {
        const unsigned bufA = sbase + (unsigned)(buf * A_REG) * 2;
        const unsigned bufB = sbase + (unsigned)(2 * A_REG + buf * B_REG) * 2;
        // A hi / lo
        const size_t ga = (size_t)(mblk + sAr) * KDIM + kk + sAc;
        const unsigned la = (unsigned)(sAr * ASTR + sAc) * 2;
        async_ld_b128(bufA + (oAh0 - 0) * 0 + la, xh + ga);          // A-hi
        async_ld_b128(bufA + (unsigned)oAl0 * 2 - (unsigned)oAh0 * 0 + la - la + (unsigned)(64 * ASTR) * 2 + (unsigned)(sAr * ASTR + sAc) * 2 - (unsigned)(sAr * ASTR + sAc) * 2 + la, xl + ga); // A-lo
        // B hi / lo
#pragma unroll
        for (int t = 0; t < 4; ++t) {
            const size_t gb = (size_t)sBcol[t] * KDIM + kk + sBc[t];
            const unsigned lb = (unsigned)(sBr[t] * ASTR + sBc[t]) * 2;
            async_ld_b128(bufB + lb, wh + gb);
            async_ld_b128(bufB + (unsigned)(256 * ASTR) * 2 + lb, wl + gb);
        }
    };

    // fragment LDS indices (halves, relative to buffer region starts)
    const int mr     = lane & 15;
    const int khalfA = (lane >> 4) * 8;
    const int khalfB = (lane >> 4) * 8;
    int aRow[2], bRow[4];
#pragma unroll
    for (int mt = 0; mt < 2; ++mt) aRow[mt] = waveM * 32 + mt * 16 + mr;
#pragma unroll
    for (int nf = 0; nf < 4; ++nf) bRow[nf] = waveN * 64 + nf * 16 + mr;

    v8f acc[2][4] = {};

    stage(0, 0);
    wait_async0();
    __syncthreads();

    int buf = 0;
    for (int kk = 0; kk < KDIM; kk += 32) {
        const bool more = (kk + 32) < KDIM;
        if (more) stage(buf ^ 1, kk + 32);

        const bf16* Abuf = smem + (size_t)buf * A_REG;
        const bf16* Bbuf = smem + (size_t)(2 * A_REG + buf * B_REG);

        FragBF Ah[2], Al[2], Bh[4], Bl[4];
#pragma unroll
        for (int mt = 0; mt < 2; ++mt) {
            const int o = aRow[mt] * ASTR + khalfA;
            Ah[mt].h[0] = *(const v8bf*)(Abuf + o);
            Ah[mt].h[1] = *(const v8bf*)(Abuf + o + 16);
            Al[mt].h[0] = *(const v8bf*)(Abuf + 64 * ASTR + o);
            Al[mt].h[1] = *(const v8bf*)(Abuf + 64 * ASTR + o + 16);
        }
#pragma unroll
        for (int nf = 0; nf < 4; ++nf) {
            const int o = bRow[nf] * ASTR + khalfB;
            Bh[nf].h[0] = *(const v8bf*)(Bbuf + o);
            Bh[nf].h[1] = *(const v8bf*)(Bbuf + o + 16);
            Bl[nf].h[0] = *(const v8bf*)(Bbuf + 256 * ASTR + o);
            Bl[nf].h[1] = *(const v8bf*)(Bbuf + 256 * ASTR + o + 16);
        }
#pragma unroll
        for (int mt = 0; mt < 2; ++mt)
#pragma unroll
            for (int nf = 0; nf < 4; ++nf) {
                acc[mt][nf] = __builtin_amdgcn_wmma_f32_16x16x32_bf16(
                    false, Ah[mt].v, false, Bh[nf].v, (short)0, acc[mt][nf], false, false);
                acc[mt][nf] = __builtin_amdgcn_wmma_f32_16x16x32_bf16(
                    false, Ah[mt].v, false, Bl[nf].v, (short)0, acc[mt][nf], false, false);
                acc[mt][nf] = __builtin_amdgcn_wmma_f32_16x16x32_bf16(
                    false, Al[mt].v, false, Bh[nf].v, (short)0, acc[mt][nf], false, false);
            }

        if (more) {
            wait_async0();
            __syncthreads();
        }
        buf ^= 1;
    }

    // epilogue: C layout lane<16 -> M=m0+v, N=lane; lane>=16 -> M=m0+8+v, N=lane-16
    const int nl = lane & 15;
    const int mh = (lane >> 4) * 8;
#pragma unroll
    for (int mt = 0; mt < 2; ++mt)
#pragma unroll
        for (int nf = 0; nf < 4; ++nf) {
            const int n = nblk + waveN * 64 + nf * 16 + nl;
            if (n < NCOLS) {
                const int mrow = mblk + waveM * 32 + mt * 16 + mh;
                bf16* outp = logits + (size_t)mrow * NCOLS + n;
#pragma unroll
                for (int v = 0; v < 8; ++v)
                    outp[(size_t)v * NCOLS] = (bf16)acc[mt][nf][v];
            }
        }
}

// ---------------------------------------------------------------- 3) BN stats, deterministic two-stage
__global__ void ndt_stats_partial(const bf16* __restrict__ logits,
                                  float* __restrict__ psum, float* __restrict__ psq)
{
    const int n = blockIdx.x * 256 + threadIdx.x;
    const int chunk = blockIdx.y;            // 0..7, 256 rows each
    if (n >= NCOLS) return;
    float s = 0.f, s2 = 0.f;
    const int mb = chunk * 256;
    for (int m = mb; m < mb + 256; ++m) {
        float v = (float)logits[(size_t)m * NCOLS + n];
        s += v; s2 += v * v;
    }
    psum[(size_t)chunk * NCOLS + n] = s;
    psq [(size_t)chunk * NCOLS + n] = s2;
}

__global__ void ndt_stats_final(const float* __restrict__ psum, const float* __restrict__ psq,
                                const float* __restrict__ gamma, const float* __restrict__ beta,
                                float* __restrict__ scale, float* __restrict__ shift)
{
    const int n = blockIdx.x * 256 + threadIdx.x;
    if (n >= NCOLS) return;
    float s = 0.f, s2 = 0.f;
#pragma unroll
    for (int c = 0; c < 8; ++c) { s += psum[(size_t)c * NCOLS + n]; s2 += psq[(size_t)c * NCOLS + n]; }
    const float mean = s * (1.f / (float)BATCH);
    const float var  = s2 * (1.f / (float)BATCH) - mean * mean;
    const float sc   = gamma[n] * rsqrtf(var + EPS_BN);
    scale[n] = sc;
    shift[n] = beta[n] - mean * sc;
}

// ---------------------------------------------------------------- 4) soft decision trees
#define SROWS 16       // samples per block
#define LSTR  1028     // LDS row stride (halves): 2056 B == 8 mod 256 -> conflict-free lane reads

__global__ __launch_bounds__(256, 1)
void ndt_tree_kernel(const bf16* __restrict__ logits, const float* __restrict__ scale,
                     const float* __restrict__ shift, float* __restrict__ part)
{
    __shared__ __align__(16) bf16 sl[SROWS * LSTR];   // 32,896 B
    __shared__ float ssc[1024];
    __shared__ float ssh[1024];
    __shared__ float pbuf[16][SROWS];

    const int t   = blockIdx.y;             // tree
    const int m0  = blockIdx.x * SROWS;     // sample base (grid.x = 128)
    const int tid = threadIdx.x;
    const size_t nbase = (size_t)t * FORKS;

    for (int j = tid; j < FORKS; j += 256) { ssc[j] = scale[nbase + j]; ssh[j] = shift[nbase + j]; }
    const int lane = tid & 31, wid = tid >> 5;
    for (int r = wid; r < SROWS; r += 8)
        for (int j = lane; j < FORKS; j += 32)
            sl[r * LSTR + j] = logits[(size_t)(m0 + r) * NCOLS + nbase + j];
    __syncthreads();

    const int s   = lane & 15;                  // sample within block
    const int sub = (wid << 1) | (lane >> 4);   // level-4 subtree, 0..15

    auto rv = [&](int j) -> float { return (float)sl[s * LSTR + j] * ssc[j] + ssh[j]; };
    auto sg = [&](float r) -> float { return 1.f / (1.f + __expf(-r)); };

    // ancestors: levels 0..3 down to the level-4 subtree root
    float prod = 1.f;
#pragma unroll
    for (int i = 0; i < 4; ++i) {
        const int q = sub >> (4 - i);
        const int j = (1 << i) - 1 + q;
        const float g = sg(rv(j));
        prod *= ((sub >> (3 - i)) & 1) ? g : (1.f - g);
    }

    float p1 = 0.f;
    for (int leaf = 0; leaf < 32; ++leaf) {
        const int q9 = (sub << 5) | leaf;       // level-9 (last fork level) node
        float p = prod;
#pragma unroll
        for (int L = 4; L < 9; ++L) {
            const int qc = q9 >> (8 - L);       // child node at level L+1
            const int j  = (1 << L) - 1 + (qc >> 1);
            const float g = sg(rv(j));
            p *= (qc & 1) ? g : (1.f - g);
        }
        p1 += p * sg(rv(511 + q9));             // right-branch prob at the last level
    }
    pbuf[sub][s] = p1;
    __syncthreads();

    if (tid < SROWS) {
        float tot = 0.f;
#pragma unroll
        for (int u = 0; u < 16; ++u) tot += pbuf[u][tid];
        part[(size_t)t * BATCH + m0 + tid] = tot;
    }
}

// ---------------------------------------------------------------- 5) combine trees -> capped output
__global__ void ndt_combine_kernel(const float* __restrict__ part, float* __restrict__ out)
{
    const int m = blockIdx.x * 256 + threadIdx.x;
    if (m >= BATCH) return;
    float s = 0.f;
#pragma unroll
    for (int t = 0; t < TREES; ++t) s += part[(size_t)t * BATCH + m];
    const float p1 = s * (1.f / (float)TREES);
    out[m] = (1.f - 2.f * EPS_CAP) * p1 + EPS_CAP;
}

// ---------------------------------------------------------------- launch
extern "C" void kernel_launch(void* const* d_in, const int* in_sizes, int n_in,
                              void* d_out, int out_size, void* d_ws, size_t ws_size,
                              hipStream_t stream)
{
    const float* x     = (const float*)d_in[0];
    const float* W     = (const float*)d_in[1];
    // d_in[2] = b (zeros; cancels exactly inside BatchNorm -> unused)
    const float* gamma = (const float*)d_in[3];
    const float* beta  = (const float*)d_in[4];
    float* out = (float*)d_out;

    // workspace carve-up (256 B aligned)
    size_t off = 0;
    auto alloc = [&](size_t sz) { size_t o = off; off = (off + sz + 255) & ~(size_t)255; return o; };
    const size_t o_xh  = alloc((size_t)BATCH * KDIM * 2);
    const size_t o_xl  = alloc((size_t)BATCH * KDIM * 2);
    const size_t o_wh  = alloc((size_t)NCOLS * KDIM * 2);
    const size_t o_wl  = alloc((size_t)NCOLS * KDIM * 2);
    const size_t o_log = alloc((size_t)BATCH * NCOLS * 2);
    const size_t o_ps  = alloc((size_t)8 * NCOLS * 4);
    const size_t o_pq  = alloc((size_t)8 * NCOLS * 4);
    const size_t o_sc  = alloc((size_t)NCOLS * 4);
    const size_t o_sh  = alloc((size_t)NCOLS * 4);
    const size_t o_pt  = alloc((size_t)TREES * BATCH * 4);
    if (ws_size < off) return;   // workspace too small -> cannot run

    char* ws = (char*)d_ws;
    bf16*  xh = (bf16*)(ws + o_xh);
    bf16*  xl = (bf16*)(ws + o_xl);
    bf16*  wh = (bf16*)(ws + o_wh);
    bf16*  wl = (bf16*)(ws + o_wl);
    bf16*  lg = (bf16*)(ws + o_log);
    float* ps = (float*)(ws + o_ps);
    float* pq = (float*)(ws + o_pq);
    float* sc = (float*)(ws + o_sc);
    float* sh = (float*)(ws + o_sh);
    float* pt = (float*)(ws + o_pt);

    ndt_split_kernel<<<4096, 256, 0, stream>>>(x, W, xh, xl, wh, wl);
    ndt_gemm_kernel<<<dim3(128, 32), 256, 0, stream>>>(xh, xl, wh, wl, lg);
    ndt_stats_partial<<<dim3(128, 8), 256, 0, stream>>>(lg, ps, pq);
    ndt_stats_final<<<128, 256, 0, stream>>>(ps, pq, gamma, beta, sc, sh);
    ndt_tree_kernel<<<dim3(BATCH / SROWS, TREES), 256, 0, stream>>>(lg, sc, sh, pt);
    ndt_combine_kernel<<<(BATCH + 255) / 256, 256, 0, stream>>>(pt, out);
}
